// LSTM_26972394619494
// MI455X (gfx1250) — matrix operand order repaired and checked by reading, not compiled, not run on hardware
//
#include <hip/hip_runtime.h>
#include <hip/hip_bf16.h>

// LSTM twin-tower model on MI455X (gfx1250, wave32).
// Pipeline:
//   k_cast   : W_ih, W_hh fp32 -> bf16 copies in workspace
//   k_gather : X[s][t][:] = bf16(emb[token_s[t]][:])
//   k_init   : zero h double-buffers + grid-barrier state (every call)
//   k_gemm   : Gpre[s][t][j] = sum_e X[s][t][e]*W_ih[j][e]   (v_wmma_f32_16x16x32_bf16)
//   k_lstm   : persistent recurrence, 32 blocks/seq, atomic grid barrier per step
//   k_head   : h1*h2 -> FC(2) -> softmax over singleton axis (== ones)
// Workspace requirement: ~96 MB (see offsets below).

#define TSEQ 4096
#define EMB  1024
#define HID  1024
#define G4   4096   // 4*HID

typedef __attribute__((ext_vector_type(16))) __bf16 bfx16;
typedef __attribute__((ext_vector_type(8)))  float  fx8;

// ---- workspace layout (bytes) ----
#define WIH_OFF   0ull
#define WHH_OFF   (WIH_OFF + (size_t)G4 * EMB * 2)          //  +8 MB
#define XBF_OFF   (WHH_OFF + (size_t)G4 * HID * 2)          //  +8 MB
#define GPRE_OFF  (XBF_OFF + 2ull * TSEQ * EMB * 2)         // +16 MB
#define HBUF_OFF  (GPRE_OFF + 2ull * TSEQ * G4 * 2)         // +64 MB
#define BAR_OFF   (HBUF_OFF + 2ull * 2 * HID * 4)           // +16 KB

// ---------------------------------------------------------------- k_cast
__global__ __launch_bounds__(256) void k_cast(const float* __restrict__ Wih,
                                              const float* __restrict__ Whh,
                                              __hip_bfloat16* __restrict__ wih,
                                              __hip_bfloat16* __restrict__ whh) {
    const size_t Nw = (size_t)G4 * EMB;
    size_t stride = (size_t)gridDim.x * blockDim.x;
    for (size_t i = (size_t)blockIdx.x * blockDim.x + threadIdx.x; i < 2 * Nw; i += stride) {
        if (i < Nw) wih[i] = __float2bfloat16(Wih[i]);
        else        whh[i - Nw] = __float2bfloat16(Whh[i - Nw]);
    }
}

// ---------------------------------------------------------------- k_gather
__global__ __launch_bounds__(256) void k_gather(const int* __restrict__ tok1,
                                                const int* __restrict__ tok2,
                                                const float* __restrict__ emb,
                                                __hip_bfloat16* __restrict__ X) {
    const int t = blockIdx.x;
    const int s = blockIdx.y;
    const int row = (s ? tok2 : tok1)[t];
    const float* src = emb + (size_t)row * EMB;
    __hip_bfloat16* dst = X + ((size_t)s * TSEQ + t) * EMB;
    for (int e = threadIdx.x; e < EMB; e += blockDim.x)
        dst[e] = __float2bfloat16(src[e]);
}

// ---------------------------------------------------------------- k_init
__global__ void k_init(float* __restrict__ hbuf, int* __restrict__ bar) {
    for (int i = threadIdx.x; i < 2 * 2 * HID; i += blockDim.x) hbuf[i] = 0.f;
    if (threadIdx.x < 32) bar[threadIdx.x] = 0;
}

// ---------------------------------------------------------------- k_gemm
// Gpre = X(T x 1024) * W_ih^T -> (T x 4096), bf16 in / fp32 acc / bf16 out.
// Block: 256 threads = 8 waves; wave tile 16(M) x 64(N); block tile 32 x 256.
__global__ __launch_bounds__(256) void k_gemm(const __hip_bfloat16* __restrict__ X,
                                              const __hip_bfloat16* __restrict__ W,
                                              __hip_bfloat16* __restrict__ Gp) {
    const int s    = blockIdx.z;
    const int tid  = threadIdx.x;
    const int lane = tid & 31;
    const int wave = tid >> 5;
    const int t0 = blockIdx.y * 32 + (wave >> 2) * 16;
    const int j0 = blockIdx.x * 256 + (wave & 3) * 64;

    const __hip_bfloat16* Xs = X + (size_t)s * TSEQ * EMB;

    // A (16x32 bf16) per-lane layout: row M = lane%16; lanes 0-15 hold
    // K=0..15, lanes 16-31 K=16..31.
    // B (32x16 bf16): col N = lane%16; lanes 0-15 hold K=0..15, lanes 16-31 K=16..31.
    const int m  = lane & 15;
    const int kA = (lane < 16) ? 0 : 16;
    const int kB = (lane < 16) ? 0 : 16;

    const __hip_bfloat16* Arow = Xs + (size_t)(t0 + m) * EMB + kA;
    const __hip_bfloat16* B0 = W + (size_t)(j0 +  0 + m) * EMB + kB;
    const __hip_bfloat16* B1 = W + (size_t)(j0 + 16 + m) * EMB + kB;
    const __hip_bfloat16* B2 = W + (size_t)(j0 + 32 + m) * EMB + kB;
    const __hip_bfloat16* B3 = W + (size_t)(j0 + 48 + m) * EMB + kB;

    fx8 acc0 = {}, acc1 = {}, acc2 = {}, acc3 = {};
    union V16 { bfx16 v; uint4 u[2]; };

    for (int e0 = 0; e0 < EMB; e0 += 32) {
        V16 a, b;
        a.u[0] = *(const uint4*)(Arow + e0);        // K elems 0..7
        a.u[1] = *(const uint4*)(Arow + e0 + 8);    // K elems 8..15

        b.u[0] = *(const uint4*)(B0 + e0);
        b.u[1] = *(const uint4*)(B0 + e0 + 8);
        acc0 = __builtin_amdgcn_wmma_f32_16x16x32_bf16(false, a.v, false, b.v, (short)0, acc0, false, false);

        b.u[0] = *(const uint4*)(B1 + e0);
        b.u[1] = *(const uint4*)(B1 + e0 + 8);
        acc1 = __builtin_amdgcn_wmma_f32_16x16x32_bf16(false, a.v, false, b.v, (short)0, acc1, false, false);

        b.u[0] = *(const uint4*)(B2 + e0);
        b.u[1] = *(const uint4*)(B2 + e0 + 8);
        acc2 = __builtin_amdgcn_wmma_f32_16x16x32_bf16(false, a.v, false, b.v, (short)0, acc2, false, false);

        b.u[0] = *(const uint4*)(B3 + e0);
        b.u[1] = *(const uint4*)(B3 + e0 + 8);
        acc3 = __builtin_amdgcn_wmma_f32_16x16x32_bf16(false, a.v, false, b.v, (short)0, acc3, false, false);
    }

    // C/D layout: VGPR r -> M = r (lanes 0-15) or 8+r (lanes 16-31); N = lane%16.
    const int mb = (lane < 16) ? 0 : 8;
    const int n  = lane & 15;
    const size_t base = (size_t)s * TSEQ * G4;
#pragma unroll
    for (int r = 0; r < 8; ++r) {
        const size_t row = base + (size_t)(t0 + mb + r) * G4 + j0 + n;
        Gp[row +  0] = __float2bfloat16(acc0[r]);
        Gp[row + 16] = __float2bfloat16(acc1[r]);
        Gp[row + 32] = __float2bfloat16(acc2[r]);
        Gp[row + 48] = __float2bfloat16(acc3[r]);
    }
}

// ---------------------------------------------------------------- k_lstm
// grid = (32, 2): 32 blocks per sequence, blockIdx.y = sequence.
// Block owns 32 hidden units n; each n gets 8 lanes for the K=1024 dots of
// all four gate rows (n, n+1024, n+2048, n+3072). c lives in registers.
__global__ __launch_bounds__(256) void k_lstm(const __hip_bfloat16* __restrict__ Whh,
                                              const __hip_bfloat16* __restrict__ Gp,
                                              const float* __restrict__ b_ih,
                                              const float* __restrict__ b_hh,
                                              float* __restrict__ hbuf,
                                              int* __restrict__ bar) {
    const int s   = blockIdx.y;
    const int tid = threadIdx.x;
    const int nl  = tid >> 3;      // 0..31 hidden unit within block
    const int l8  = tid & 7;       // 0..7 lane within the dot-group
    const int n   = blockIdx.x * 32 + nl;

    const float bias0 = b_ih[n]           + b_hh[n];
    const float bias1 = b_ih[HID + n]     + b_hh[HID + n];
    const float bias2 = b_ih[2 * HID + n] + b_hh[2 * HID + n];
    const float bias3 = b_ih[3 * HID + n] + b_hh[3 * HID + n];

    const __hip_bfloat16* W0 = Whh + (size_t)(n) * HID;
    const __hip_bfloat16* W1 = Whh + (size_t)(HID + n) * HID;
    const __hip_bfloat16* W2 = Whh + (size_t)(2 * HID + n) * HID;
    const __hip_bfloat16* W3 = Whh + (size_t)(3 * HID + n) * HID;

    const __hip_bfloat16* Gs = Gp + (size_t)s * TSEQ * G4;
    float* hb = hbuf + (size_t)s * 2 * HID;
    int* cnt = bar + s * 16;       // separate cache lines per sequence
    int* gen = bar + s * 16 + 1;

    __shared__ float hs[HID];
    float c = 0.f;

    for (int t = 0; t < TSEQ; ++t) {
        const float* hin  = hb + (t & 1) * HID;
        float*       hout = hb + ((t & 1) ^ 1) * HID;

        for (int i = tid; i < HID; i += 256) hs[i] = hin[i];
        __syncthreads();

        float a0 = 0.f, a1 = 0.f, a2 = 0.f, a3 = 0.f;
        for (int it = 0; it < 16; ++it) {
            const int k = it * 64 + l8 * 8;
            if (it < 15) {                             // L2-resident rows -> L0
                __builtin_prefetch(W0 + k + 64, 0, 0); // global_prefetch_b8
                __builtin_prefetch(W1 + k + 64, 0, 0);
                __builtin_prefetch(W2 + k + 64, 0, 0);
                __builtin_prefetch(W3 + k + 64, 0, 0);
            }
            float h8[8];
#pragma unroll
            for (int e = 0; e < 8; ++e) h8[e] = hs[k + e];
            uint4 w0 = *(const uint4*)(W0 + k);
            uint4 w1 = *(const uint4*)(W1 + k);
            uint4 w2 = *(const uint4*)(W2 + k);
            uint4 w3 = *(const uint4*)(W3 + k);
            const __hip_bfloat16* p0 = (const __hip_bfloat16*)&w0;
            const __hip_bfloat16* p1 = (const __hip_bfloat16*)&w1;
            const __hip_bfloat16* p2 = (const __hip_bfloat16*)&w2;
            const __hip_bfloat16* p3 = (const __hip_bfloat16*)&w3;
#pragma unroll
            for (int e = 0; e < 8; ++e) {
                a0 = fmaf(__bfloat162float(p0[e]), h8[e], a0);
                a1 = fmaf(__bfloat162float(p1[e]), h8[e], a1);
                a2 = fmaf(__bfloat162float(p2[e]), h8[e], a2);
                a3 = fmaf(__bfloat162float(p3[e]), h8[e], a3);
            }
        }
        // reduce across the 8 lanes of this n (wave32 subgroups of 8)
#pragma unroll
        for (int d = 4; d > 0; d >>= 1) {
            a0 += __shfl_down(a0, d, 8);
            a1 += __shfl_down(a1, d, 8);
            a2 += __shfl_down(a2, d, 8);
            a3 += __shfl_down(a3, d, 8);
        }

        if (l8 == 0) {
            const __hip_bfloat16* g = Gs + (size_t)t * G4;
            const float gi = a0 + __bfloat162float(g[n])           + bias0;
            const float gf = a1 + __bfloat162float(g[HID + n])     + bias1;
            const float gg = a2 + __bfloat162float(g[2 * HID + n]) + bias2;
            const float go = a3 + __bfloat162float(g[3 * HID + n]) + bias3;
            const float iv = 1.f / (1.f + __expf(-gi));
            const float fv = 1.f / (1.f + __expf(-gf));
            const float gv = tanhf(gg);
            const float ov = 1.f / (1.f + __expf(-go));
            c = fv * c + iv * gv;
            hout[n] = ov * tanhf(c);
        }

        // grid barrier over this sequence's 32 blocks (sense-reversing).
        __syncthreads();
        if (tid == 0) {
            __threadfence();
            const int gcur = __hip_atomic_load(gen, __ATOMIC_ACQUIRE, __HIP_MEMORY_SCOPE_AGENT);
            const int prev = __hip_atomic_fetch_add(cnt, 1, __ATOMIC_ACQ_REL, __HIP_MEMORY_SCOPE_AGENT);
            if (prev == 31) {
                __hip_atomic_store(cnt, 0, __ATOMIC_RELAXED, __HIP_MEMORY_SCOPE_AGENT);
                __hip_atomic_fetch_add(gen, 1, __ATOMIC_ACQ_REL, __HIP_MEMORY_SCOPE_AGENT);
            } else {
                while (__hip_atomic_load(gen, __ATOMIC_ACQUIRE, __HIP_MEMORY_SCOPE_AGENT) == gcur) {
                    __builtin_amdgcn_s_sleep(1);
                }
            }
        }
        __syncthreads();
    }
}

// ---------------------------------------------------------------- k_head
__global__ __launch_bounds__(256) void k_head(const float* __restrict__ hbuf,
                                              const float* __restrict__ W_fc,
                                              const float* __restrict__ b_fc,
                                              float* __restrict__ out) {
    __shared__ float r0[256], r1[256];
    const int tid = threadIdx.x;
    const float* h1 = hbuf;                 // seq 0, final parity 0
    const float* h2 = hbuf + 2 * HID;       // seq 1, final parity 0
    float p0 = 0.f, p1 = 0.f;
    for (int n = tid; n < HID; n += 256) {
        const float hf = h1[n] * h2[n];     // combo_unit == 'MULT'
        p0 = fmaf(hf, W_fc[n], p0);
        p1 = fmaf(hf, W_fc[HID + n], p1);
    }
    r0[tid] = p0; r1[tid] = p1;
    __syncthreads();
    for (int s = 128; s > 0; s >>= 1) {
        if (tid < s) { r0[tid] += r0[tid + s]; r1[tid] += r1[tid + s]; }
        __syncthreads();
    }
    if (tid == 0) {
        const float l0 = r0[0] + b_fc[0];
        const float l1 = r1[0] + b_fc[1];
        // Faithful to reference: softmax over dim=1 (singleton axis) -> ones.
        out[0] = __expf(l0 - l0);
        out[1] = __expf(l1 - l1);
    }
}

// ---------------------------------------------------------------- launch
extern "C" void kernel_launch(void* const* d_in, const int* in_sizes, int n_in,
                              void* d_out, int out_size, void* d_ws, size_t ws_size,
                              hipStream_t stream) {
    const int*   tok1 = (const int*)d_in[0];
    const int*   tok2 = (const int*)d_in[1];
    const float* emb  = (const float*)d_in[2];
    const float* W_ih = (const float*)d_in[3];
    const float* W_hh = (const float*)d_in[4];
    const float* b_ih = (const float*)d_in[5];
    const float* b_hh = (const float*)d_in[6];
    const float* W_fc = (const float*)d_in[7];
    const float* b_fc = (const float*)d_in[8];

    char* ws = (char*)d_ws;
    __hip_bfloat16* wih  = (__hip_bfloat16*)(ws + WIH_OFF);
    __hip_bfloat16* whh  = (__hip_bfloat16*)(ws + WHH_OFF);
    __hip_bfloat16* xbf  = (__hip_bfloat16*)(ws + XBF_OFF);
    __hip_bfloat16* gpre = (__hip_bfloat16*)(ws + GPRE_OFF);
    float* hbuf = (float*)(ws + HBUF_OFF);
    int*   bar  = (int*)(ws + BAR_OFF);

    k_cast  <<<4096, 256, 0, stream>>>(W_ih, W_hh, wih, whh);
    k_gather<<<dim3(TSEQ, 2), 256, 0, stream>>>(tok1, tok2, emb, xbf);
    k_init  <<<1, 256, 0, stream>>>(hbuf, bar);
    k_gemm  <<<dim3(16, 128, 2), 256, 0, stream>>>(xbf, wih, gpre);
    k_lstm  <<<dim3(32, 2), 256, 0, stream>>>(whh, gpre, b_ih, b_hh, hbuf, bar);
    k_head  <<<1, 256, 0, stream>>>(hbuf, W_fc, b_fc, (float*)d_out);
}
